// FlaxLoRACrossFrameAttention_76184129896669
// MI455X (gfx1250) — matrix operand order repaired and checked by reading, not compiled, not run on hardware
//
#include <hip/hip_runtime.h>
#include <stdint.h>
#include <stddef.h>

// ---------------------------------------------------------------------------
// FlaxLoRACrossFrameAttention on MI455X (gfx1250, wave32, WMMA bf16)
//
// Working set fits in the 192MB L2 -> matrix-pipe bound. Use
// V_WMMA_F32_16X16X32_BF16 with fp32 accumulation; LoRA (rank 4) folded into
// the weights. GEMMs are register-blocked 32x64 per wave; the shared weight
// tile is staged into LDS with GLOBAL_LOAD_ASYNC_TO_LDS_B128 (double
// buffered, ASYNCcnt) when the builtin is available.
// ---------------------------------------------------------------------------

#define D_MODEL 512
#define N_TOK   1024
#define BF_CNT  32      // batch * frames
#define NHEADS  8
#define DHEAD   64
#define TOKENS  (BF_CNT * N_TOK)   // 32768

#if defined(__has_builtin)
#if __has_builtin(__builtin_amdgcn_global_load_async_to_lds_b128) && \
    __has_builtin(__builtin_amdgcn_s_wait_asynccnt)
#define HAVE_ASYNC_LDS 1
#endif
#endif
#ifndef HAVE_ASYNC_LDS
#define HAVE_ASYNC_LDS 0
#endif

typedef __attribute__((ext_vector_type(16))) __bf16 v16bf;
typedef __attribute__((ext_vector_type(8)))  float  v8f;
typedef __attribute__((ext_vector_type(4)))  int    v4i;

union FragU { v16bf v; uint4 q[2]; };

#define WMMA_BF16(a, b, c) \
  __builtin_amdgcn_wmma_f32_16x16x32_bf16(false, (a), false, (b), (short)0, (c), false, false)

// round-to-nearest-even f32 -> bf16 (bit pattern as ushort)
__device__ __forceinline__ unsigned short f2bf(float f) {
  union { float f; unsigned u; } x; x.f = f;
  unsigned r = x.u + 0x7FFFu + ((x.u >> 16) & 1u);
  return (unsigned short)(r >> 16);
}

// Load a 16x(2x8) bf16 fragment as two 16-byte chunks (A-frag / B-frag / LDS).
__device__ __forceinline__ v16bf load_frag(const unsigned short* p0,
                                           const unsigned short* p1) {
  FragU u;
  u.q[0] = *reinterpret_cast<const uint4*>(p0);
  u.q[1] = *reinterpret_cast<const uint4*>(p1);
  return u.v;
}

// ---------------------------------------------------------------------------
// Kernel 1: fold LoRA into weight, store transposed bf16:  WT[d][c]
//   eff = W[c][d] + sum_r down[c][r] * up[r][d]     (LORA_SCALE == 1)
// ---------------------------------------------------------------------------
__global__ void prep_w(const float* __restrict__ W, const float* __restrict__ dn,
                       const float* __restrict__ up, unsigned short* __restrict__ WT) {
  int e = blockIdx.x * 256 + threadIdx.x;     // 0 .. 512*512-1
  int c = e >> 9;
  int d = e & 511;
  float acc = W[c * 512 + d];
#pragma unroll
  for (int r = 0; r < 4; ++r) acc += dn[c * 4 + r] * up[r * 512 + d];
  WT[(size_t)d * 512 + c] = f2bf(acc);
}

// ---------------------------------------------------------------------------
// Kernel 2: fp32 activations -> bf16
// ---------------------------------------------------------------------------
__global__ void cvt_hs(const float* __restrict__ x, unsigned short* __restrict__ y) {
  size_t i = ((size_t)blockIdx.x * 256 + threadIdx.x) * 4;
  float4 v = *reinterpret_cast<const float4*>(x + i);
  uint2 p;
  p.x = (unsigned)f2bf(v.x) | ((unsigned)f2bf(v.y) << 16);
  p.y = (unsigned)f2bf(v.z) | ((unsigned)f2bf(v.w) << 16);
  *reinterpret_cast<uint2*>(y + i) = p;
}

// ---------------------------------------------------------------------------
// Shared GEMM core: workgroup = 8 waves covering 256(M) x 64(N); each wave
// owns a 32x64 block (8 fp32 WMMA accumulators). The 64x32 weight tile for
// the current k-step is staged into LDS once per workgroup, double buffered,
// via async load-to-LDS when available.
// ---------------------------------------------------------------------------
__device__ __forceinline__ void stage_b(const unsigned short* __restrict__ WT,
                                        unsigned short* __restrict__ buf,
                                        int n0, int kk, int tid) {
  int row = tid >> 2;            // 0..63   (output column within N tile)
  int ch  = (tid & 3) * 8;       // element offset 0,8,16,24 (16B chunks)
  const unsigned short* g = WT + (size_t)(n0 + row) * 512 + kk + ch;
  unsigned short* l = buf + row * 32 + ch;
#if HAVE_ASYNC_LDS
  __builtin_amdgcn_global_load_async_to_lds_b128(
      (__attribute__((address_space(1))) v4i*)g,
      (__attribute__((address_space(3))) v4i*)l, 0, 0);
#else
  *reinterpret_cast<uint4*>(l) = *reinterpret_cast<const uint4*>(g);
#endif
}

__device__ __forceinline__ void gemm_core(const unsigned short* __restrict__ A,
                                          const unsigned short* __restrict__ WT,
                                          unsigned short* __restrict__ lds,  // 2*2048 elems
                                          int m0, int n0, int tid,
                                          v8f acc[2][4]) {
  int lane = tid & 31;
  int half = lane >> 4, l15 = lane & 15;
  int kbase = half * 8;

  const unsigned short* Arow0 = A + (size_t)(m0 + l15) * 512;
  const unsigned short* Arow1 = Arow0 + (size_t)16 * 512;

  stage_b(WT, lds, n0, 0, tid);        // prologue: buffer 0

  for (int it = 0; it < 16; ++it) {
    int kk = it * 32;
    unsigned short* cur = lds + (it & 1) * 2048;
#if HAVE_ASYNC_LDS
    __builtin_amdgcn_s_wait_asynccnt(0);
#endif
    __syncthreads();                   // staged tile visible to all waves
    if (it + 1 < 16) stage_b(WT, lds + ((it + 1) & 1) * 2048, n0, kk + 32, tid);

    __builtin_prefetch(Arow0 + kk + 128, 0, 0);
    __builtin_prefetch(Arow1 + kk + 128, 0, 0);
    v16bf a0 = load_frag(Arow0 + kk + kbase, Arow0 + kk + kbase + 16);
    v16bf a1 = load_frag(Arow1 + kk + kbase, Arow1 + kk + kbase + 16);
#pragma unroll
    for (int j = 0; j < 4; ++j) {
      const unsigned short* bp = cur + (j * 16 + l15) * 32 + half * 8;
      v16bf bfr = load_frag(bp, bp + 16);
      acc[0][j] = WMMA_BF16(a0, bfr, acc[0][j]);
      acc[1][j] = WMMA_BF16(a1, bfr, acc[1][j]);
    }
  }
}

// ---------------------------------------------------------------------------
// Kernel 3: QKV projection GEMM.  C[32768x512] = A[32768x512] @ W_eff
// grid.x: 128 M-blocks x 8 N-blocks; blockIdx.z: 0->q, 1->k, 2->v(transposed)
// ---------------------------------------------------------------------------
__global__ void gemm_qkv(const unsigned short* __restrict__ A,
                         const unsigned short* __restrict__ wqT,
                         const unsigned short* __restrict__ wkT,
                         const unsigned short* __restrict__ wvT,
                         unsigned short* __restrict__ qo,
                         unsigned short* __restrict__ ko,
                         unsigned short* __restrict__ vo) {
  __shared__ unsigned short bstage[2 * 2048];

  int tid = threadIdx.x;
  int lane = tid & 31, wave = tid >> 5;
  int half = lane >> 4, l15 = lane & 15;
  int m0 = (blockIdx.x >> 3) * 256 + wave * 32;
  int n0 = (blockIdx.x & 7) * 64;

  const unsigned short* WT = (blockIdx.z == 0) ? wqT : (blockIdx.z == 1) ? wkT : wvT;

  v8f acc[2][4] = {};
  gemm_core(A, WT, bstage, m0, n0, tid, acc);

  if (blockIdx.z < 2) {
    unsigned short* Out = (blockIdx.z == 0) ? qo : ko;
#pragma unroll
    for (int mi = 0; mi < 2; ++mi)
#pragma unroll
      for (int j = 0; j < 4; ++j) {
        int col = n0 + j * 16 + l15;
#pragma unroll
        for (int r = 0; r < 8; ++r) {
          int token = m0 + mi * 16 + r + 8 * half;
          Out[(size_t)token * 512 + col] = f2bf(acc[mi][j][r]);
        }
      }
  } else {
    // V stored transposed per head: v_t[bf][h][dh][n]  -> contiguous PV B-frags
#pragma unroll
    for (int mi = 0; mi < 2; ++mi)
#pragma unroll
      for (int j = 0; j < 4; ++j) {
        int col = n0 + j * 16 + l15;
        int hh = col >> 6, dh = col & 63;
#pragma unroll
        for (int r = 0; r < 8; ++r) {
          int token = m0 + mi * 16 + r + 8 * half;
          int bfi = token >> 10, nn = token & 1023;
          vo[(((size_t)bfi * NHEADS + hh) * DHEAD + dh) * 1024 + nn] = f2bf(acc[mi][j][r]);
        }
      }
  }
}

// ---------------------------------------------------------------------------
// Kernel 4: cross-frame flash attention.
// grid.x = BF*H (256) head-batches, grid.y = 8 query tiles of 128.
// Each wave: 16 queries, online softmax over 1024 keys in 32-key chunks.
// K/V come from frame max(f-1, 0)  (the cross-frame gather is index math).
// ---------------------------------------------------------------------------
__global__ void attn(const unsigned short* __restrict__ Q,
                     const unsigned short* __restrict__ K,
                     const unsigned short* __restrict__ VT,
                     unsigned short* __restrict__ H) {
  __shared__ unsigned short pbuf[8 * 16 * 32];   // per-wave 16x32 bf16 P tile

  int lane = threadIdx.x & 31, wave = threadIdx.x >> 5;
  int half = lane >> 4, l15 = lane & 15;
  int kbase = half * 8;

  int hb = blockIdx.x;               // 0..255
  int bf = hb >> 3, h = hb & 7;
  int b = bf >> 4, f = bf & 15;      // BATCH=2, F=16
  int pf = (f == 0) ? 0 : (f - 1);   // previous-frame gather
  int kvbf = (b << 4) | pf;
  int q0 = blockIdx.y * 128 + wave * 16;

  // Q A-fragments for dh 0..31 and 32..63 (hoisted out of the key loop)
  const unsigned short* qrow = Q + ((size_t)(bf * 1024 + q0 + l15)) * 512 + h * 64;
  v16bf aq0 = load_frag(qrow + kbase,      qrow + kbase + 16);
  v16bf aq1 = load_frag(qrow + 32 + kbase, qrow + 32 + kbase + 16);

  float rm[8], rl[8];
  v8f o0 = {}, o1 = {}, o2 = {}, o3 = {};
#pragma unroll
  for (int r = 0; r < 8; ++r) { rm[r] = -1e30f; rl[r] = 0.0f; }

  unsigned short* pw = pbuf + wave * 512;
  const unsigned short* kbasep = K + ((size_t)kvbf * 1024) * 512 + h * 64 + half * 8;
  const unsigned short* vbasep = VT + (((size_t)kvbf * NHEADS + h) * DHEAD) * 1024;

  for (int n0 = 0; n0 < 1024; n0 += 32) {
    // ---- S = (Q @ K^T) * 1/sqrt(dh) : two 16x16 tiles, K-dim = 64 (2 WMMAs each)
    v8f s0 = {}, s1 = {};
    const unsigned short* kr0 = kbasep + (size_t)(n0 + l15) * 512;       // keys n0..n0+15
    const unsigned short* kr1 = kr0 + (size_t)16 * 512;                  // keys n0+16..n0+31
    {
      v16bf b0 = load_frag(kr0, kr0 + 16);
      v16bf b1 = load_frag(kr1, kr1 + 16);
      s0 = WMMA_BF16(aq0, b0, s0);
      s1 = WMMA_BF16(aq0, b1, s1);
    }
    {
      v16bf b0 = load_frag(kr0 + 32, kr0 + 48);
      v16bf b1 = load_frag(kr1 + 32, kr1 + 48);
      s0 = WMMA_BF16(aq1, b0, s0);
      s1 = WMMA_BF16(aq1, b1, s1);
    }

    // ---- online softmax (row = r + 8*half, columns live across 16 lanes)
    float p0v[8], p1v[8];
#pragma unroll
    for (int r = 0; r < 8; ++r) {
      float a = s0[r] * 0.125f;
      float c = s1[r] * 0.125f;
      float mx = fmaxf(a, c);
#pragma unroll
      for (int m = 1; m < 16; m <<= 1) mx = fmaxf(mx, __shfl_xor(mx, m, 32));
      float mn = fmaxf(rm[r], mx);
      float alpha = __expf(rm[r] - mn);
      rm[r] = mn;
      float e0 = __expf(a - mn);
      float e1 = __expf(c - mn);
      p0v[r] = e0; p1v[r] = e1;
      float rs = e0 + e1;
#pragma unroll
      for (int m = 1; m < 16; m <<= 1) rs += __shfl_xor(rs, m, 32);
      rl[r] = rl[r] * alpha + rs;
      o0[r] *= alpha; o1[r] *= alpha; o2[r] *= alpha; o3[r] *= alpha;
    }

    // ---- transpose P (C-layout -> A-layout) through LDS, as bf16
#pragma unroll
    for (int r = 0; r < 8; ++r) {
      int m = r + 8 * half;
      pw[m * 32 + l15]      = f2bf(p0v[r]);
      pw[m * 32 + 16 + l15] = f2bf(p1v[r]);
    }
    v16bf ap = load_frag(pw + l15 * 32 + kbase, pw + l15 * 32 + kbase + 16);

    // ---- O += P @ V : 4 dh-column groups of 16
    const unsigned short* vb = vbasep + n0 + half * 8;
    {
      const unsigned short* p = vb + (size_t)(0 * 16 + l15) * 1024;
      o0 = WMMA_BF16(ap, load_frag(p, p + 16), o0);
    }
    {
      const unsigned short* p = vb + (size_t)(1 * 16 + l15) * 1024;
      o1 = WMMA_BF16(ap, load_frag(p, p + 16), o1);
    }
    {
      const unsigned short* p = vb + (size_t)(2 * 16 + l15) * 1024;
      o2 = WMMA_BF16(ap, load_frag(p, p + 16), o2);
    }
    {
      const unsigned short* p = vb + (size_t)(3 * 16 + l15) * 1024;
      o3 = WMMA_BF16(ap, load_frag(p, p + 16), o3);
    }
  }

  // ---- normalize and write h back to channel layout (bf16 for the out-proj GEMM)
#pragma unroll
  for (int r = 0; r < 8; ++r) {
    float inv = 1.0f / rl[r];
    int token = q0 + r + 8 * half;
    unsigned short* hp = H + ((size_t)(bf * 1024 + token)) * 512 + h * 64;
    hp[l15]      = f2bf(o0[r] * inv);
    hp[16 + l15] = f2bf(o1[r] * inv);
    hp[32 + l15] = f2bf(o2[r] * inv);
    hp[48 + l15] = f2bf(o3[r] * inv);
  }
}

// ---------------------------------------------------------------------------
// Kernel 5: output projection GEMM with bias, fp32 output (Wo already has
// the o_down@o_up LoRA folded in by prep_w).
// ---------------------------------------------------------------------------
__global__ void out_gemm(const unsigned short* __restrict__ A,
                         const unsigned short* __restrict__ woT,
                         const float* __restrict__ bo,
                         float* __restrict__ Out) {
  __shared__ unsigned short bstage[2 * 2048];

  int tid = threadIdx.x;
  int lane = tid & 31, wave = tid >> 5;
  int half = lane >> 4, l15 = lane & 15;
  int m0 = (blockIdx.x >> 3) * 256 + wave * 32;
  int n0 = (blockIdx.x & 7) * 64;

  v8f acc[2][4] = {};
  gemm_core(A, woT, bstage, m0, n0, tid, acc);

#pragma unroll
  for (int mi = 0; mi < 2; ++mi)
#pragma unroll
    for (int j = 0; j < 4; ++j) {
      int col = n0 + j * 16 + l15;
      float bias = bo[col];
#pragma unroll
      for (int r = 0; r < 8; ++r) {
        int token = m0 + mi * 16 + r + 8 * half;
        Out[(size_t)token * 512 + col] = acc[mi][j][r] + bias;
      }
    }
}

// ---------------------------------------------------------------------------
// Host launcher
// ---------------------------------------------------------------------------
extern "C" void kernel_launch(void* const* d_in, const int* in_sizes, int n_in,
                              void* d_out, int out_size, void* d_ws, size_t ws_size,
                              hipStream_t stream) {
  const float* hs     = (const float*)d_in[0];
  const float* Wq     = (const float*)d_in[1];
  const float* Wk     = (const float*)d_in[2];
  const float* Wv     = (const float*)d_in[3];
  const float* q_down = (const float*)d_in[4];
  const float* q_up   = (const float*)d_in[5];
  const float* k_down = (const float*)d_in[6];
  const float* k_up   = (const float*)d_in[7];
  const float* v_down = (const float*)d_in[8];
  const float* v_up   = (const float*)d_in[9];
  const float* Wo     = (const float*)d_in[10];
  const float* bo     = (const float*)d_in[11];
  const float* o_down = (const float*)d_in[12];
  const float* o_up   = (const float*)d_in[13];
  float* out = (float*)d_out;

  const size_t ACT_BYTES = (size_t)TOKENS * D_MODEL * 2;   // 33,554,432
  const size_t W_BYTES   = (size_t)D_MODEL * D_MODEL * 2;  //    524,288

  char* ws = (char*)d_ws;
  unsigned short* hs_bf = (unsigned short*)(ws + 0 * ACT_BYTES);
  unsigned short* q_bf  = (unsigned short*)(ws + 1 * ACT_BYTES);
  unsigned short* k_bf  = (unsigned short*)(ws + 2 * ACT_BYTES);
  unsigned short* v_t   = (unsigned short*)(ws + 3 * ACT_BYTES);
  unsigned short* h_bf  = (unsigned short*)(ws + 4 * ACT_BYTES);
  unsigned short* wqT   = (unsigned short*)(ws + 5 * ACT_BYTES);
  unsigned short* wkT   = (unsigned short*)(ws + 5 * ACT_BYTES + 1 * W_BYTES);
  unsigned short* wvT   = (unsigned short*)(ws + 5 * ACT_BYTES + 2 * W_BYTES);
  unsigned short* woT   = (unsigned short*)(ws + 5 * ACT_BYTES + 3 * W_BYTES);

  // Fold LoRA into the four weight matrices (transposed bf16).
  prep_w<<<1024, 256, 0, stream>>>(Wq, q_down, q_up, wqT);
  prep_w<<<1024, 256, 0, stream>>>(Wk, k_down, k_up, wkT);
  prep_w<<<1024, 256, 0, stream>>>(Wv, v_down, v_up, wvT);
  prep_w<<<1024, 256, 0, stream>>>(Wo, o_down, o_up, woT);

  // Activations fp32 -> bf16 (16,777,216 elems, 4 per thread).
  cvt_hs<<<16384, 256, 0, stream>>>(hs, hs_bf);

  // QKV projections: 128 M-blocks x 8 N-blocks, z picks q/k/v.
  gemm_qkv<<<dim3(1024, 1, 3), 256, 0, stream>>>(hs_bf, wqT, wkT, wvT,
                                                 q_bf, k_bf, v_t);

  // Cross-frame flash attention: 256 head-batches x 8 query tiles of 128.
  attn<<<dim3(256, 8), 256, 0, stream>>>(q_bf, k_bf, v_t, h_bf);

  // Output projection with bias, fp32 result.
  out_gemm<<<1024, 256, 0, stream>>>(h_bf, woT, bo, out);
}
